// CoattentionModel_15908558864876
// MI455X (gfx1250) — matrix operand order repaired and checked by reading, not compiled, not run on hardware
//
#include <hip/hip_runtime.h>
#include <hip/hip_bf16.h>

// ---------------------------------------------------------------------------
// Types for CDNA5 WMMA
// ---------------------------------------------------------------------------
typedef __attribute__((ext_vector_type(16))) __bf16 v16bf;
typedef __attribute__((ext_vector_type(8)))  float  v8f;

enum { ACT_NONE = 0, ACT_TANH = 1, ACT_RELU = 2 };

__device__ __forceinline__ unsigned short f32_to_bf16_raw(float f) {
  union { float f; unsigned u; } v; v.f = f;
  unsigned r = v.u + 0x7FFFu + ((v.u >> 16) & 1u);   // round-to-nearest-even
  return (unsigned short)(r >> 16);
}

// ---------------------------------------------------------------------------
// Generic batched WMMA GEMM:
//   C[b] = act( (accumulate? C[b] : 0) + op(A[b]) @ op(B[b]) + bias + D[b] )
// fp32 operands staged to LDS as bf16: A as [BM][32], B transposed to [BN][32]
// so both per-lane fragment gathers are contiguous in K (b128 LDS loads).
// Hot staging path uses float4 global loads; scalar fallback covers tails,
// transposes and row-shifted (conv) edges.  Wave32; each wave owns FMxFN
// 16x16 tiles computed with v_wmma_f32_16x16x32_bf16.
// A and B fragments both use contiguous K order: element s of lane half h holds k = 16h + s.
// ---------------------------------------------------------------------------
template <int BM, int BN, int WAVES_M, int WAVES_N>
__global__ __launch_bounds__(256)
void gemm_wmma(const float* __restrict__ A, long long sA, int lda,
               const float* __restrict__ Bm, long long sB, int ldb,
               float*       __restrict__ C, long long sC, int ldc,
               const float* __restrict__ bias,
               const float* __restrict__ D, long long sD, int ldd,
               int M, int N, int K,
               int rowOff, int transA, int transB,
               int accumulate, int act)
{
  constexpr int BK  = 32;
  constexpr int KV4 = BK / 4;       // float4 chunks per K-row
  constexpr int NV4 = BN / 4;       // float4 chunks per N-row
  constexpr int WM  = BM / WAVES_M;
  constexpr int WN  = BN / WAVES_N;
  constexpr int FM  = WM / 16;
  constexpr int FN  = WN / 16;

  __shared__ unsigned short As[BM][BK];   // [m][k]
  __shared__ unsigned short Bs[BN][BK];   // [n][k]  (transposed stage)

  const int batch = blockIdx.z;
  A  += (long long)batch * sA;
  Bm += (long long)batch * sB;
  C  += (long long)batch * sC;
  if (D) D += (long long)batch * sD;

  const int tid   = threadIdx.x;
  const int lane  = tid & 31;
  const int wave  = tid >> 5;
  const int wm_id = wave % WAVES_M;
  const int wn_id = wave / WAVES_M;

  const int blockM = blockIdx.y * BM;
  const int blockN = blockIdx.x * BN;

  const bool ldaV = (lda & 3) == 0;
  const bool ldbV = (ldb & 3) == 0;

  v8f acc[FM][FN];
#pragma unroll
  for (int i = 0; i < FM; ++i)
#pragma unroll
    for (int j = 0; j < FN; ++j) {
      v8f z = {0.f,0.f,0.f,0.f,0.f,0.f,0.f,0.f};
      acc[i][j] = z;
    }

  const int half = (lane >> 4) & 1;
  const int lr   = lane & 15;

  const int kTiles = (K + BK - 1) / BK;
  for (int kt = 0; kt < kTiles; ++kt) {
    const int k0 = kt * BK;

    // ---- stage A tile: 4 bf16 per thread-iteration ----
    for (int idx = tid; idx < BM * KV4; idx += 256) {
      const int m  = idx / KV4;
      const int kq = (idx % KV4) * 4;
      const int gm = blockM + m + rowOff;
      const int gk = k0 + kq;
      float f0 = 0.f, f1 = 0.f, f2 = 0.f, f3 = 0.f;
      if (gm >= 0 && gm < M) {
        if (!transA) {
          const float* src = A + (long long)gm * lda + gk;
          if (gk + 3 < K && ldaV) {
            const float4 v4 = *(const float4*)src;
            f0 = v4.x; f1 = v4.y; f2 = v4.z; f3 = v4.w;
          } else {
            if (gk + 0 < K) f0 = src[0];
            if (gk + 1 < K) f1 = src[1];
            if (gk + 2 < K) f2 = src[2];
            if (gk + 3 < K) f3 = src[3];
          }
        } else {
          if (gk + 0 < K) f0 = A[(long long)(gk + 0) * lda + gm];
          if (gk + 1 < K) f1 = A[(long long)(gk + 1) * lda + gm];
          if (gk + 2 < K) f2 = A[(long long)(gk + 2) * lda + gm];
          if (gk + 3 < K) f3 = A[(long long)(gk + 3) * lda + gm];
        }
      }
      union { unsigned short s[4]; unsigned u[2]; } pk;
      pk.s[0] = f32_to_bf16_raw(f0); pk.s[1] = f32_to_bf16_raw(f1);
      pk.s[2] = f32_to_bf16_raw(f2); pk.s[3] = f32_to_bf16_raw(f3);
      unsigned* d = (unsigned*)&As[m][kq];
      d[0] = pk.u[0]; d[1] = pk.u[1];
    }

    // ---- stage B tile into [n][k] ----
    if (transB) {
      // B stored [N,K]: contiguous in k, same pattern as A
      for (int idx = tid; idx < BN * KV4; idx += 256) {
        const int n  = idx / KV4;
        const int kq = (idx % KV4) * 4;
        const int gn = blockN + n;
        const int gk = k0 + kq;
        float f0 = 0.f, f1 = 0.f, f2 = 0.f, f3 = 0.f;
        if (gn < N) {
          const float* src = Bm + (long long)gn * ldb + gk;
          if (gk + 3 < K && ldbV) {
            const float4 v4 = *(const float4*)src;
            f0 = v4.x; f1 = v4.y; f2 = v4.z; f3 = v4.w;
          } else {
            if (gk + 0 < K) f0 = src[0];
            if (gk + 1 < K) f1 = src[1];
            if (gk + 2 < K) f2 = src[2];
            if (gk + 3 < K) f3 = src[3];
          }
        }
        union { unsigned short s[4]; unsigned u[2]; } pk;
        pk.s[0] = f32_to_bf16_raw(f0); pk.s[1] = f32_to_bf16_raw(f1);
        pk.s[2] = f32_to_bf16_raw(f2); pk.s[3] = f32_to_bf16_raw(f3);
        unsigned* d = (unsigned*)&Bs[n][kq];
        d[0] = pk.u[0]; d[1] = pk.u[1];
      }
    } else {
      // B stored [K,N]: float4 over n, scatter 4 rows of Bs
      for (int idx = tid; idx < BK * NV4; idx += 256) {
        const int k  = idx / NV4;
        const int nq = (idx % NV4) * 4;
        const int gk = k0 + k;
        const int gn = blockN + nq;
        float f0 = 0.f, f1 = 0.f, f2 = 0.f, f3 = 0.f;
        if (gk < K) {
          const float* src = Bm + (long long)gk * ldb + gn;
          if (gn + 3 < N && ldbV) {
            const float4 v4 = *(const float4*)src;
            f0 = v4.x; f1 = v4.y; f2 = v4.z; f3 = v4.w;
          } else {
            if (gn + 0 < N) f0 = src[0];
            if (gn + 1 < N) f1 = src[1];
            if (gn + 2 < N) f2 = src[2];
            if (gn + 3 < N) f3 = src[3];
          }
        }
        Bs[nq + 0][k] = f32_to_bf16_raw(f0);
        Bs[nq + 1][k] = f32_to_bf16_raw(f1);
        Bs[nq + 2][k] = f32_to_bf16_raw(f2);
        Bs[nq + 3][k] = f32_to_bf16_raw(f3);
      }
    }
    __syncthreads();

    // ---- fragments + WMMA ----
    v16bf afrag[FM], bfrag[FN];
#pragma unroll
    for (int i = 0; i < FM; ++i) {
      const int m  = wm_id * WM + i * 16 + lr;
      const int kb = half * 16;
      union { unsigned u[8]; v16bf v; } cv;
      const unsigned* rowp = (const unsigned*)&As[m][0];
#pragma unroll
      for (int p = 0; p < 8; ++p) cv.u[p] = rowp[(kb >> 1) + p];            // K=kb..kb+15
      afrag[i] = cv.v;
    }
#pragma unroll
    for (int j = 0; j < FN; ++j) {
      const int n  = wn_id * WN + j * 16 + lr;
      const int kb = half * 16;
      union { unsigned u[8]; v16bf v; } cv;
      const unsigned* rowp = (const unsigned*)&Bs[n][0];
#pragma unroll
      for (int p = 0; p < 8; ++p) cv.u[p] = rowp[(kb >> 1) + p];            // K=kb..kb+15
      bfrag[j] = cv.v;
    }
#pragma unroll
    for (int i = 0; i < FM; ++i)
#pragma unroll
      for (int j = 0; j < FN; ++j)
        acc[i][j] = __builtin_amdgcn_wmma_f32_16x16x32_bf16(
            false, afrag[i], false, bfrag[j], (short)0, acc[i][j], false, false);

    __syncthreads();
  }

  // ---- epilogue ----
#pragma unroll
  for (int i = 0; i < FM; ++i) {
#pragma unroll
    for (int j = 0; j < FN; ++j) {
      const int gn = blockN + wn_id * WN + j * 16 + lr;
#pragma unroll
      for (int r = 0; r < 8; ++r) {
        const int gm = blockM + wm_id * WM + i * 16 + r + half * 8;
        if (gm < M && gn < N) {
          float v = acc[i][j][r];
          long long cidx = (long long)gm * ldc + gn;
          if (accumulate) v += C[cidx];
          if (bias)       v += bias[gn];
          if (D)          v += D[(long long)gm * ldd + gn];
          if (act == ACT_TANH)      v = tanhf(v);
          else if (act == ACT_RELU) v = fmaxf(v, 0.0f);
          C[cidx] = v;
        }
      }
    }
  }
}

// ---------------------------------------------------------------------------
// Pointwise / glue kernels
// ---------------------------------------------------------------------------
__global__ void zero_kernel(float* p, int n) {
  int i = blockIdx.x * blockDim.x + threadIdx.x;
  if (i < n) p[i] = 0.0f;
}

__global__ void embed_kernel(const int* __restrict__ q,
                             const float* __restrict__ table,
                             float* __restrict__ words) {
  int row = blockIdx.x;                       // b*T + t
  int idx = q[row];
  const float4* src = (const float4*)(table + (long long)idx * 512);
  float4*       dst = (float4*)(words + (long long)row * 512);
  dst[threadIdx.x] = src[threadIdx.x];        // blockDim = 128 (512 floats)
}

__global__ void phrase_kernel(const float* __restrict__ u,
                              const float* __restrict__ b2,
                              const float* __restrict__ tr,
                              const float* __restrict__ bu,
                              const float* __restrict__ bb,
                              const float* __restrict__ bt,
                              float* __restrict__ out, int n) {
  int i = blockIdx.x * blockDim.x + threadIdx.x;
  if (i >= n) return;
  int h = i & 511;
  float a = tanhf(u[i]  + bu[h]);
  float c = tanhf(b2[i] + bb[h]);
  float d = tanhf(tr[i] + bt[h]);
  out[i] = tanhf(fmaxf(a, fmaxf(c, d)));
}

__global__ void lstm_gate(const float* __restrict__ zg,   // [256,2048] = h@rk
                          const float* __restrict__ xpt, int ldxp,
                          float* __restrict__ h, float* __restrict__ c,
                          float* __restrict__ out_h, int ldout) {
  int idx = blockIdx.x * blockDim.x + threadIdx.x;        // b*512 + hh
  if (idx >= 256 * 512) return;
  int b = idx >> 9, hh = idx & 511;
  const float* zb = zg + (long long)b * 2048;
  const float* xb = xpt + (long long)b * ldxp;
  float gi = zb[hh]        + xb[hh];
  float gf = zb[512 + hh]  + xb[512 + hh];
  float gg = zb[1024 + hh] + xb[1024 + hh];
  float go = zb[1536 + hh] + xb[1536 + hh];
  gi = 1.f / (1.f + __expf(-gi));
  gf = 1.f / (1.f + __expf(-gf));
  go = 1.f / (1.f + __expf(-go));
  float cn = gf * c[idx] + gi * tanhf(gg);
  float hn = go * tanhf(cn);
  c[idx] = cn; h[idx] = hn;
  out_h[(long long)b * ldout + hh] = hn;
}

// scores = softmax_r( sum[b,r,:].w + bias ); qv[b,:] (+)= sum_r scores[r]*feat[b,r,:]
__global__ __launch_bounds__(256)
void attention_kernel(const float* __restrict__ sum, long long sSum,
                      const float* __restrict__ feat, long long sFeat,
                      const float* __restrict__ w, const float* __restrict__ bsc,
                      float* __restrict__ qv, int ldqv, int rows, int accumulate) {
  __shared__ float sc[256];
  const int b = blockIdx.x;
  const float* S = sum  + (long long)b * sSum;
  const float* F = feat + (long long)b * sFeat;
  const int tid = threadIdx.x, lane = tid & 31, wave = tid >> 5;
  const float bias = bsc[0];
  for (int r = wave; r < rows; r += 8) {
    float p = 0.f;
    const float* row = S + (long long)r * 512;
    for (int i = lane; i < 512; i += 32) p += row[i] * w[i];
#pragma unroll
    for (int off = 16; off > 0; off >>= 1) p += __shfl_xor(p, off, 32);
    if (lane == 0) sc[r] = p + bias;
  }
  __syncthreads();
  if (tid == 0) {
    float mx = -1e30f;
    for (int r = 0; r < rows; ++r) mx = fmaxf(mx, sc[r]);
    float s = 0.f;
    for (int r = 0; r < rows; ++r) { float e = __expf(sc[r] - mx); sc[r] = e; s += e; }
    float inv = 1.f / s;
    for (int r = 0; r < rows; ++r) sc[r] *= inv;
  }
  __syncthreads();
  for (int h = tid; h < 512; h += 256) {
    float a = 0.f;
    for (int r = 0; r < rows; ++r) a += sc[r] * F[(long long)r * 512 + h];
    float* o = qv + (long long)b * ldqv + h;
    if (accumulate) *o += a; else *o = a;
  }
}

__global__ __launch_bounds__(256)
void softmax_kernel(const float* __restrict__ logits, float* __restrict__ out, int C) {
  __shared__ float red[256];
  int b = blockIdx.x, tid = threadIdx.x;
  const float* L = logits + (long long)b * C;
  float mx = -1e30f;
  for (int i = tid; i < C; i += 256) mx = fmaxf(mx, L[i]);
  red[tid] = mx; __syncthreads();
  for (int s = 128; s > 0; s >>= 1) { if (tid < s) red[tid] = fmaxf(red[tid], red[tid + s]); __syncthreads(); }
  mx = red[0]; __syncthreads();
  float sm = 0.f;
  for (int i = tid; i < C; i += 256) sm += __expf(L[i] - mx);
  red[tid] = sm; __syncthreads();
  for (int s = 128; s > 0; s >>= 1) { if (tid < s) red[tid] += red[tid + s]; __syncthreads(); }
  float inv = 1.f / red[0];
  for (int i = tid; i < C; i += 256) out[(long long)b * C + i] = __expf(L[i] - mx) * inv;
}

// ---------------------------------------------------------------------------
// Host-side launch helpers
// ---------------------------------------------------------------------------
static void gemmL(hipStream_t st, const float* A, long long sA, int lda,
                  const float* Bm, long long sB, int ldb,
                  float* C, long long sC, int ldc,
                  const float* bias, const float* D, long long sD, int ldd,
                  int M, int N, int K, int rowOff, int tA, int tB,
                  int accum, int act, int batch) {
  dim3 grid((N + 127) / 128, (M + 127) / 128, batch);
  gemm_wmma<128, 128, 4, 2><<<grid, dim3(256), 0, st>>>(
      A, sA, lda, Bm, sB, ldb, C, sC, ldc, bias, D, sD, ldd,
      M, N, K, rowOff, tA, tB, accum, act);
}
static void gemmS(hipStream_t st, const float* A, long long sA, int lda,
                  const float* Bm, long long sB, int ldb,
                  float* C, long long sC, int ldc,
                  const float* bias, const float* D, long long sD, int ldd,
                  int M, int N, int K, int rowOff, int tA, int tB,
                  int accum, int act, int batch) {
  dim3 grid((N + 63) / 64, (M + 31) / 32, batch);
  gemm_wmma<32, 64, 2, 4><<<grid, dim3(256), 0, st>>>(
      A, sA, lda, Bm, sB, ldb, C, sC, ldc, bias, D, sD, ldd,
      M, N, K, rowOff, tA, tB, accum, act);
}

// ---------------------------------------------------------------------------
// kernel_launch
// ---------------------------------------------------------------------------
extern "C" void kernel_launch(void* const* d_in, const int* in_sizes, int n_in,
                              void* d_out, int out_size, void* d_ws, size_t ws_size,
                              hipStream_t stream) {
  constexpr int kB = 256, kN = 196, kT = 26, kH = 512, kC = 1000;
  constexpr int kBN = kB * kN;     // 50176
  constexpr int kBT = kB * kT;     // 6656

  const float* image    = (const float*)d_in[0];
  const int*   question = (const int*)  d_in[1];
  const float* table    = (const float*)d_in[2];
  const float* W_img = (const float*)d_in[3];  const float* b_img = (const float*)d_in[4];
  const float* W_uni = (const float*)d_in[5];  const float* b_uni = (const float*)d_in[6];
  const float* W_bi  = (const float*)d_in[7];  const float* b_bi  = (const float*)d_in[8];
  const float* W_tri = (const float*)d_in[9];  const float* b_tri = (const float*)d_in[10];
  const float* lstm_k  = (const float*)d_in[11];
  const float* lstm_rk = (const float*)d_in[12];
  const float* lstm_b  = (const float*)d_in[13];
  const float* W_corr = (const float*)d_in[14]; const float* b_corr = (const float*)d_in[15];
  const float* W_qe   = (const float*)d_in[16]; const float* b_qe   = (const float*)d_in[17];
  const float* W_ie   = (const float*)d_in[18]; const float* b_ie   = (const float*)d_in[19];
  const float* w_qa   = (const float*)d_in[20]; const float* b_qa   = (const float*)d_in[21];
  const float* w_ia   = (const float*)d_in[22]; const float* b_ia   = (const float*)d_in[23];
  const float* W_w    = (const float*)d_in[24]; const float* b_w    = (const float*)d_in[25];
  const float* W_p    = (const float*)d_in[26]; const float* b_p    = (const float*)d_in[27];
  const float* W_s    = (const float*)d_in[28]; const float* b_s    = (const float*)d_in[29];
  const float* W_fc1  = (const float*)d_in[30]; const float* b_fc1  = (const float*)d_in[31];
  const float* W_fc   = (const float*)d_in[32]; const float* b_fc   = (const float*)d_in[33];

  (void)in_sizes; (void)n_in; (void)out_size; (void)ws_size;

  // ---- workspace bump allocator ----
  char* wsp = (char*)d_ws;
  auto alloc = [&](size_t nfloats) -> float* {
    float* p = (float*)wsp;
    wsp += ((nfloats * sizeof(float)) + 255) & ~(size_t)255;
    return p;
  };
  float* img      = alloc((size_t)kBN * kH);
  float* img_corr = alloc((size_t)kBN * kH);
  float* img_emb  = alloc((size_t)kBN * kH);
  float* img_sum  = alloc((size_t)kBN * kH);
  float* words    = alloc((size_t)kBT * kH);
  float* uni      = alloc((size_t)kBT * kH);
  float* bi2      = alloc((size_t)kBT * kH);
  float* tri      = alloc((size_t)kBT * kH);
  float* phrase   = alloc((size_t)kBT * kH);
  float* sent     = alloc((size_t)kBT * kH);
  float* xp       = alloc((size_t)kBT * 4 * kH);
  float* zg       = alloc((size_t)kB * 4 * kH);
  float* hbuf     = alloc((size_t)kB * kH);
  float* cbuf     = alloc((size_t)kB * kH);
  float* qemb     = alloc((size_t)kBT * kH);
  float* qsum     = alloc((size_t)kBT * kH);
  float* wm       = alloc((size_t)kB * kT * kN);
  float* qv_w     = alloc((size_t)kB * kH);
  float* cat1     = alloc((size_t)kB * 2 * kH);   // [qv_phr | h_w]
  float* cat2     = alloc((size_t)kB * 2 * kH);   // [qv_sent | h_p]
  float* hs_buf   = alloc((size_t)kB * kH);
  float* fc1      = alloc((size_t)kB * 2 * kH);
  float* logits   = alloc((size_t)kB * kC);

  // 1) img = tanh(image @ W_img + b_img)
  gemmL(stream, image, 0, kH, W_img, 0, kH, img, 0, kH, b_img, nullptr, 0, 0,
        kBN, kH, kH, 0, 0, 0, 0, ACT_TANH, 1);

  // 2) words = embed_table[question]
  embed_kernel<<<kBT, 128, 0, stream>>>(question, table, words);

  // 3) dilated convs as shifted batched GEMMs (raw, biases applied in pointwise)
  const long long sQ = (long long)kT * kH;
  gemmS(stream, words, sQ, kH, W_uni,            0, kH, uni, sQ, kH, nullptr, nullptr, 0, 0,
        kT, kH, kH,  0, 0, 0, 0, ACT_NONE, kB);
  gemmS(stream, words, sQ, kH, W_bi,             0, kH, bi2, sQ, kH, nullptr, nullptr, 0, 0,
        kT, kH, kH, -1, 0, 0, 0, ACT_NONE, kB);
  gemmS(stream, words, sQ, kH, W_bi + kH * kH,   0, kH, bi2, sQ, kH, nullptr, nullptr, 0, 0,
        kT, kH, kH, +1, 0, 0, 1, ACT_NONE, kB);
  gemmS(stream, words, sQ, kH, W_tri,            0, kH, tri, sQ, kH, nullptr, nullptr, 0, 0,
        kT, kH, kH, -2, 0, 0, 0, ACT_NONE, kB);
  gemmS(stream, words, sQ, kH, W_tri + kH * kH,  0, kH, tri, sQ, kH, nullptr, nullptr, 0, 0,
        kT, kH, kH,  0, 0, 0, 1, ACT_NONE, kB);
  gemmS(stream, words, sQ, kH, W_tri + 2*kH*kH,  0, kH, tri, sQ, kH, nullptr, nullptr, 0, 0,
        kT, kH, kH, +2, 0, 0, 1, ACT_NONE, kB);
  phrase_kernel<<<(kBT * kH + 255) / 256, 256, 0, stream>>>(
      uni, bi2, tri, b_uni, b_bi, b_tri, phrase, kBT * kH);

  // 4) LSTM: xp = phrase @ lstm_k + lstm_b
  gemmL(stream, phrase, 0, kH, lstm_k, 0, 4 * kH, xp, 0, 4 * kH, lstm_b, nullptr, 0, 0,
        kBT, 4 * kH, kH, 0, 0, 0, 0, ACT_NONE, 1);
  zero_kernel<<<(kB * kH + 255) / 256, 256, 0, stream>>>(hbuf, kB * kH);
  zero_kernel<<<(kB * kH + 255) / 256, 256, 0, stream>>>(cbuf, kB * kH);
  for (int t = 0; t < kT; ++t) {
    gemmL(stream, hbuf, 0, kH, lstm_rk, 0, 4 * kH, zg, 0, 4 * kH, nullptr, nullptr, 0, 0,
          kB, 4 * kH, kH, 0, 0, 0, 0, ACT_NONE, 1);
    lstm_gate<<<(kB * kH + 255) / 256, 256, 0, stream>>>(
        zg, xp + (long long)t * 4 * kH, kT * 4 * kH, hbuf, cbuf,
        sent + (long long)t * kH, kT * kH);
  }

  // 5) co-attention shared image projections
  gemmL(stream, img, 0, kH, W_corr, 0, kH, img_corr, 0, kH, b_corr, nullptr, 0, 0,
        kBN, kH, kH, 0, 0, 0, 0, ACT_NONE, 1);
  gemmL(stream, img, 0, kH, W_ie, 0, kH, img_emb, 0, kH, b_ie, nullptr, 0, 0,
        kBN, kH, kH, 0, 0, 0, 0, ACT_NONE, 1);

  const float* qfeat[3]  = { words, phrase, sent };
  float*       qvtgt[3]  = { qv_w, cat1, cat2 };
  const int    ldqv[3]   = { kH, 2 * kH, 2 * kH };
  const long long sI  = (long long)kN * kH;
  const long long sWm = (long long)kT * kN;

  for (int s = 0; s < 3; ++s) {
    const float* q = qfeat[s];
    // ques_embed = q @ W_qe + b_qe
    gemmL(stream, q, 0, kH, W_qe, 0, kH, qemb, 0, kH, b_qe, nullptr, 0, 0,
          kBT, kH, kH, 0, 0, 0, 0, ACT_NONE, 1);
    // wm = tanh(q @ img_corr^T)     [B, T, N]
    gemmS(stream, q, sQ, kH, img_corr, sI, kH, wm, sWm, kN, nullptr, nullptr, 0, 0,
          kT, kN, kH, 0, 0, 1, 0, ACT_TANH, kB);
    // ques_sum = tanh(wm @ img_embed + ques_embed)   [B, T, H]
    gemmS(stream, wm, sWm, kN, img_emb, sI, kH, qsum, sQ, kH, nullptr, qemb, sQ, kH,
          kT, kH, kN, 0, 0, 0, 0, ACT_TANH, kB);
    // img_sum = tanh(wm^T @ ques_embed + img_embed)  [B, N, H]
    gemmS(stream, wm, sWm, kN, qemb, sQ, kH, img_sum, sI, kH, nullptr, img_emb, sI, kH,
          kN, kH, kT, 0, 1, 0, 0, ACT_TANH, kB);
    // attention + weighted feature sums; qv = q_feat + v_feat
    attention_kernel<<<kB, 256, 0, stream>>>(qsum, sQ, q, sQ, w_qa, b_qa,
                                             qvtgt[s], ldqv[s], kT, 0);
    attention_kernel<<<kB, 256, 0, stream>>>(img_sum, sI, img, sI, w_ia, b_ia,
                                             qvtgt[s], ldqv[s], kN, 1);
  }

  // 6) head: h_w -> cat1 cols [H,2H); h_p -> cat2 cols [H,2H); h_s; fc
  gemmL(stream, qv_w, 0, kH, W_w, 0, kH, cat1 + kH, 0, 2 * kH, b_w, nullptr, 0, 0,
        kB, kH, kH, 0, 0, 0, 0, ACT_TANH, 1);
  gemmL(stream, cat1, 0, 2 * kH, W_p, 0, kH, cat2 + kH, 0, 2 * kH, b_p, nullptr, 0, 0,
        kB, kH, 2 * kH, 0, 0, 0, 0, ACT_TANH, 1);
  gemmL(stream, cat2, 0, 2 * kH, W_s, 0, kH, hs_buf, 0, kH, b_s, nullptr, 0, 0,
        kB, kH, 2 * kH, 0, 0, 0, 0, ACT_TANH, 1);
  gemmL(stream, hs_buf, 0, kH, W_fc1, 0, 2 * kH, fc1, 0, 2 * kH, b_fc1, nullptr, 0, 0,
        kB, 2 * kH, kH, 0, 0, 0, 0, ACT_RELU, 1);
  gemmL(stream, fc1, 0, 2 * kH, W_fc, 0, kC, logits, 0, kC, b_fc, nullptr, 0, 0,
        kB, kC, 2 * kH, 0, 0, 0, 0, ACT_NONE, 1);

  softmax_kernel<<<kB, 256, 0, stream>>>(logits, (float*)d_out, kC);
}
